// DiagonalSSMBlock_62302795596617
// MI455X (gfx1250) — compile-verified, matrix operand order checked by reading, not run on hardware
//
#include <hip/hip_runtime.h>

// ---------------------------------------------------------------------------
// DiagonalSSMBlock for MI455X (gfx1250, wave32, WMMA + TDM)
//   x:[4,4096,1024] f32 -> out f32, D_FF=2736 (padded to 2752 for WMMA K/N)
// ---------------------------------------------------------------------------

typedef __attribute__((ext_vector_type(16))) __bf16        v16bf;
typedef __attribute__((ext_vector_type(8)))  float         v8f;
typedef __attribute__((ext_vector_type(4)))  unsigned int  u32x4;
typedef __attribute__((ext_vector_type(4)))  unsigned int  uint32x4_t;
typedef __attribute__((ext_vector_type(8)))  int           int32x8_t;
typedef __attribute__((ext_vector_type(4)))  int           int32x4_t;
typedef unsigned short u16;

#define D_MODEL 1024
#define N_STATE 256
#define T_SEQ   4096
#define B_SZ    4
#define M_ROWS  (B_SZ * T_SEQ)   // 16384
#define D_FF    2736
#define D_FF_P  2752             // 86*32, 43*64
#define EPS_RMS 1e-6f

#if __has_builtin(__builtin_amdgcn_tensor_load_to_lds)
#define HAVE_TDM 1
#else
#define HAVE_TDM 0
#endif
#if defined(__clang_major__) && (__clang_major__ >= 23)
#define TDM_6ARG 1
#else
#define TDM_6ARG 0
#endif

__device__ __forceinline__ u16 f2bf(float f) {
  unsigned int u = __float_as_uint(f);
  unsigned int r = (u + 0x7fffu + ((u >> 16) & 1u)) >> 16; // RNE
  return (u16)r;
}
__device__ __forceinline__ float bf2f(u16 s) {
  return __uint_as_float(((unsigned int)s) << 16);
}
__device__ __forceinline__ float sigm(float x) { return 1.f / (1.f + __expf(-x)); }

// ---------------------------------------------------------------------------
// fp32 -> bf16 weight convert with zero padding (dst [Rpad x Kpad] row-major)
// ---------------------------------------------------------------------------
__global__ __launch_bounds__(256)
void k_cvt_bf16(const float* __restrict__ src, u16* __restrict__ dst,
                int R, int K, int Kpad, int total) {
  int idx = blockIdx.x * 256 + threadIdx.x;
  if (idx >= total) return;
  int r = idx / Kpad;
  int k = idx - r * Kpad;
  float v = (r < R && k < K) ? src[(size_t)r * K + k] : 0.f;
  dst[idx] = f2bf(v);
}

// ---------------------------------------------------------------------------
// RMSNorm: one 256-thread block per row of 1024; bf16 output
// ---------------------------------------------------------------------------
__global__ __launch_bounds__(256)
void k_rmsnorm_bf16(const float* __restrict__ x, const float* __restrict__ w,
                    u16* __restrict__ o) {
  int row = blockIdx.x;
  const float* xr = x + (size_t)row * D_MODEL;
  int i0 = threadIdx.x * 4;
  float4 v = *(const float4*)(xr + i0);
  float ss = v.x * v.x + v.y * v.y + v.z * v.z + v.w * v.w;
  #pragma unroll
  for (int ofs = 16; ofs > 0; ofs >>= 1) ss += __shfl_xor(ss, ofs, 32);
  __shared__ float red[8];
  if ((threadIdx.x & 31) == 0) red[threadIdx.x >> 5] = ss;
  __syncthreads();
  float tot = 0.f;
  #pragma unroll
  for (int i = 0; i < 8; i++) tot += red[i];
  float inv = rsqrtf(tot * (1.f / (float)D_MODEL) + EPS_RMS);
  float4 wv = *(const float4*)(w + i0);
  u16* orow = o + (size_t)row * D_MODEL + i0;
  orow[0] = f2bf(v.x * inv * wv.x);
  orow[1] = f2bf(v.y * inv * wv.y);
  orow[2] = f2bf(v.z * inv * wv.z);
  orow[3] = f2bf(v.w * inv * wv.w);
}

// ---------------------------------------------------------------------------
// Chunked diagonal scan: h_t = lam*h + Bu_t   (T=4096 = 64 chunks x 64 steps)
// ---------------------------------------------------------------------------
__global__ __launch_bounds__(256)
void k_scan_chunk(const float* __restrict__ Bu, const float* __restrict__ llam,
                  float* __restrict__ csum) {
  int idx = blockIdx.x * 256 + threadIdx.x;      // [b][ch][n]
  int n = idx & (N_STATE - 1);
  int ch = (idx >> 8) & 63;
  int b = idx >> 14;
  float lam = sigm(llam[n]);
  const float* p = Bu + ((size_t)b * T_SEQ + (size_t)ch * 64) * N_STATE + n;
  float h = 0.f;
  #pragma unroll 8
  for (int i = 0; i < 64; i++) h = fmaf(lam, h, p[(size_t)i * N_STATE]);
  csum[idx] = h;
}

__global__ __launch_bounds__(256)
void k_scan_carry(const float* __restrict__ llam, const float* __restrict__ csum,
                  float* __restrict__ carry) {
  int idx = blockIdx.x * 256 + threadIdx.x;      // b*256+n (1024 total)
  int n = idx & (N_STATE - 1);
  int b = idx >> 8;
  float lam = sigm(llam[n]);
  float lamL = __powf(lam, 64.f);
  float h = 0.f;
  for (int ch = 0; ch < 64; ch++) {
    size_t o = ((size_t)b * 64 + ch) * N_STATE + n;
    carry[o] = h;
    h = fmaf(lamL, h, csum[o]);
  }
}

__global__ __launch_bounds__(256)
void k_scan_out(const float* __restrict__ Bu, const float* __restrict__ llam,
                const float* __restrict__ carry, u16* __restrict__ hs) {
  int idx = blockIdx.x * 256 + threadIdx.x;
  int n = idx & (N_STATE - 1);
  int ch = (idx >> 8) & 63;
  int b = idx >> 14;
  float lam = sigm(llam[n]);
  float h = carry[((size_t)b * 64 + ch) * N_STATE + n];
  const float* p = Bu + ((size_t)b * T_SEQ + (size_t)ch * 64) * N_STATE + n;
  u16* q = hs + ((size_t)b * T_SEQ + (size_t)ch * 64) * N_STATE + n;
  #pragma unroll 8
  for (int i = 0; i < 64; i++) {
    h = fmaf(lam, h, p[(size_t)i * N_STATE]);
    q[(size_t)i * N_STATE] = f2bf(h);
  }
}

// ---------------------------------------------------------------------------
// TDM: DMA a [rows x 32]-bf16 tile (row pitch lda elements) into LDS with
// +16-half padding per row (pad_interval=16 DWORDs, pad_amount=8 DWORDs)
// so the LDS row stride is 48 halves = LK. D# per ISA 8.3/8.4.
// ---------------------------------------------------------------------------
#if HAVE_TDM
__device__ __forceinline__ void tdm_load_tile(const u16* gsrc, const u16* lds_dst,
                                              int k_len, int lda, int rows) {
  unsigned long long ga = (unsigned long long)(size_t)(const void*)gsrc;
  unsigned la = (unsigned)(unsigned long long)(size_t)(const void*)lds_dst;
  uint32x4_t g0;
  g0[0] = 1u;                                            // count=1, user-mode load
  g0[1] = la;                                            // lds_addr
  g0[2] = (unsigned)ga;                                  // global_addr[31:0]
  g0[3] = (unsigned)((ga >> 32) & 0x01ffffffu) | (2u << 30);  // addr[56:32], type=2
  int32x8_t g1;
  g1[0] = (int)((1u << 16) |        // data_size = 2 bytes
                (1u << 20) |        // pad_enable
                (3u << 22) |        // pad_interval: 16 DWORDs
                (7u << 25));        // pad_amount: 8 DWORDs (=16 halves)
  g1[1] = (int)(((unsigned)k_len & 0xffffu) << 16);                       // tensor_dim0 lo
  g1[2] = (int)((((unsigned)k_len >> 16) & 0xffffu) |
                (((unsigned)rows & 0xffffu) << 16));                      // dim0 hi | dim1 lo
  g1[3] = (int)(32u << 16);                                               // tile_dim0 = 32
  g1[4] = (int)((unsigned)rows & 0xffffu);                                // tile_dim1 = rows
  g1[5] = lda;                                                            // tensor_dim0_stride
  g1[6] = 0;
  g1[7] = 0;
  int32x4_t z4 = {0, 0, 0, 0};
#if TDM_6ARG
  int32x8_t z8 = {0, 0, 0, 0, 0, 0, 0, 0};
  __builtin_amdgcn_tensor_load_to_lds(g0, g1, z4, z4, z8, 0);
#else
  __builtin_amdgcn_tensor_load_to_lds(g0, g1, z4, z4, 0);
#endif
}
#endif

// ---------------------------------------------------------------------------
// WMMA bf16 GEMM:  C[M,N] = A[M,K] * W[N,K]^T
//   Block tile 128x64, 8 waves (4x2), each wave 32x32 = 2x2 wmma tiles.
//   LDA/LDW/LDC compile-time so address math folds into instruction offsets.
//   TDM double-buffered staging when available; manual staging otherwise.
//   MODE 0: Cf = acc
//   MODE 1: Cf = resid + acc + dskip[c]*bf2f(U)   (SSM output + residual)
//   MODE 2: Cf = resid + acc                      (FFN output + residual)
//   MODE 3: dual accumulate W and W3, Cb = bf16(silu(acc)*acc3)  (SwiGLU)
// ---------------------------------------------------------------------------
__device__ __forceinline__ v16bf load_frag(const u16* lds, int rowbase, int hi) {
  union { v16bf v; u32x4 q[2]; } u;
  u.q[0] = *(const u32x4*)(lds + rowbase + hi * 8);
  u.q[1] = *(const u32x4*)(lds + rowbase + 16 + hi * 8);
  return u.v;
}

template <int MODE, int LDA, int LDW, int LDC>
__global__ __launch_bounds__(256)
void k_gemm_bf16(const u16* __restrict__ A,
                 const u16* __restrict__ W, const u16* __restrict__ W3,
                 float* __restrict__ Cf, u16* __restrict__ Cb, int K,
                 const float* __restrict__ resid,
                 const u16* __restrict__ U,
                 const float* __restrict__ dskip) {
  constexpr int BM = 128, BN = 64, BK = 32, LK = 48;  // LK pad keeps 16B align
#if HAVE_TDM
  constexpr int NBUF = 2;
#else
  constexpr int NBUF = 1;
#endif
  __shared__ alignas(16) u16 lA[NBUF * BM * LK];
  __shared__ alignas(16) u16 lB[NBUF * BN * LK];
  __shared__ alignas(16) u16 lB3[(MODE == 3) ? (NBUF * BN * LK) : 16];

  const int tid = threadIdx.x;
  const int wave = tid >> 5, lane = tid & 31;
  const int wm = wave >> 1, wn = wave & 1;         // 4x2 wave grid
  const int l16 = lane & 15, hi = lane >> 4;
  const int gm0 = blockIdx.y * BM, gn0 = blockIdx.x * BN;

  v8f acc[2][2] = {};
  v8f acc3[2][2] = {};

#if HAVE_TDM
  const int nk = K / BK;
  const u16* Abase = A + (size_t)gm0 * LDA;
  const u16* Wbase = W + (size_t)gn0 * LDW;
  const u16* W3base = (MODE == 3) ? (W3 + (size_t)gn0 * LDW) : nullptr;
  // prologue: stage tile 0 into buffer 0
  if (wave == 0)      tdm_load_tile(Abase, lA, K, LDA, BM);
  else if (wave == 1) tdm_load_tile(Wbase, lB, K, LDW, BN);
  else if (MODE == 3 && wave == 2) tdm_load_tile(W3base, lB3, K, LDW, BN);

  for (int k = 0; k < nk; k++) {
    const int cur = k & 1, nxt = cur ^ 1;
    if (k + 1 < nk) {
      const int k1 = (k + 1) * BK;
      if (wave == 0)      tdm_load_tile(Abase + k1, lA + nxt * (BM * LK), K, LDA, BM);
      else if (wave == 1) tdm_load_tile(Wbase + k1, lB + nxt * (BN * LK), K, LDW, BN);
      else if (MODE == 3 && wave == 2)
        tdm_load_tile(W3base + k1, lB3 + nxt * (BN * LK), K, LDW, BN);
      __builtin_amdgcn_s_wait_tensorcnt(1);   // oldest (tile k) complete
    } else {
      __builtin_amdgcn_s_wait_tensorcnt(0);
    }
    __syncthreads();                          // tile k visible to all waves

    const u16* pa  = lA  + cur * (BM * LK);
    const u16* pb  = lB  + cur * (BN * LK);
    const u16* pb3 = lB3 + cur * ((MODE == 3) ? (BN * LK) : 0);
    v16bf fa[2], fb[2], fb3[2];
    #pragma unroll
    for (int tm = 0; tm < 2; tm++)
      fa[tm] = load_frag(pa, (wm * 32 + tm * 16 + l16) * LK, hi);
    #pragma unroll
    for (int tn = 0; tn < 2; tn++)
      fb[tn] = load_frag(pb, (wn * 32 + tn * 16 + l16) * LK, hi);
    if (MODE == 3) {
      #pragma unroll
      for (int tn = 0; tn < 2; tn++)
        fb3[tn] = load_frag(pb3, (wn * 32 + tn * 16 + l16) * LK, hi);
    }
    #pragma unroll
    for (int tm = 0; tm < 2; tm++) {
      #pragma unroll
      for (int tn = 0; tn < 2; tn++) {
        acc[tm][tn] = __builtin_amdgcn_wmma_f32_16x16x32_bf16(
            false, fa[tm], false, fb[tn], (short)0, acc[tm][tn], false, false);
        if (MODE == 3)
          acc3[tm][tn] = __builtin_amdgcn_wmma_f32_16x16x32_bf16(
              false, fa[tm], false, fb3[tn], (short)0, acc3[tm][tn], false, false);
      }
    }
    __syncthreads();                          // done reading before buf reuse
  }
#else
  const int arow = tid >> 1, aseg = (tid & 1) * 16; // 128 rows x 32 halves
  const int brow = tid >> 2, bseg = (tid & 3) * 8;  // 64 rows x 32 halves
  for (int k0 = 0; k0 < K; k0 += BK) {
    __syncthreads();
    {
      const u32x4* src = (const u32x4*)(A + (size_t)(gm0 + arow) * LDA + k0 + aseg);
      *(u32x4*)(lA + arow * LK + aseg)     = src[0];
      *(u32x4*)(lA + arow * LK + aseg + 8) = src[1];
    }
    *(u32x4*)(lB + brow * LK + bseg) =
        *(const u32x4*)(W + (size_t)(gn0 + brow) * LDW + k0 + bseg);
    if (MODE == 3)
      *(u32x4*)(lB3 + brow * LK + bseg) =
          *(const u32x4*)(W3 + (size_t)(gn0 + brow) * LDW + k0 + bseg);
    __syncthreads();

    v16bf fa[2], fb[2], fb3[2];
    #pragma unroll
    for (int tm = 0; tm < 2; tm++)
      fa[tm] = load_frag(lA, (wm * 32 + tm * 16 + l16) * LK, hi);
    #pragma unroll
    for (int tn = 0; tn < 2; tn++)
      fb[tn] = load_frag(lB, (wn * 32 + tn * 16 + l16) * LK, hi);
    if (MODE == 3) {
      #pragma unroll
      for (int tn = 0; tn < 2; tn++)
        fb3[tn] = load_frag(lB3, (wn * 32 + tn * 16 + l16) * LK, hi);
    }
    #pragma unroll
    for (int tm = 0; tm < 2; tm++) {
      #pragma unroll
      for (int tn = 0; tn < 2; tn++) {
        acc[tm][tn] = __builtin_amdgcn_wmma_f32_16x16x32_bf16(
            false, fa[tm], false, fb[tn], (short)0, acc[tm][tn], false, false);
        if (MODE == 3)
          acc3[tm][tn] = __builtin_amdgcn_wmma_f32_16x16x32_bf16(
              false, fa[tm], false, fb3[tn], (short)0, acc3[tm][tn], false, false);
      }
    }
  }
#endif

  // Epilogue. C/D layout: VGPR r -> row = base + r + 8*hi, col = base + (lane&15)
  #pragma unroll
  for (int tm = 0; tm < 2; tm++) {
    #pragma unroll
    for (int tn = 0; tn < 2; tn++) {
      const int r0 = gm0 + wm * 32 + tm * 16 + hi * 8;
      const int c  = gn0 + wn * 32 + tn * 16 + l16;
      const size_t base = (size_t)r0 * LDC + c;
      #pragma unroll
      for (int r = 0; r < 8; r++) {
        const size_t o = base + (size_t)r * LDC;   // LDC constexpr -> imm offsets
        float v = acc[tm][tn][r];
        if (MODE == 0) {
          Cf[o] = v;
        } else if (MODE == 1) {
          Cf[o] = resid[o] + v + dskip[c] * bf2f(U[o]);
        } else if (MODE == 2) {
          Cf[o] = resid[o] + v;
        } else {
          float g = v * sigm(v);               // silu
          Cb[o] = f2bf(g * acc3[tm][tn][r]);
        }
      }
    }
  }
}

// ---------------------------------------------------------------------------
// Host launcher
// ---------------------------------------------------------------------------
extern "C" void kernel_launch(void* const* d_in, const int* in_sizes, int n_in,
                              void* d_out, int out_size, void* d_ws, size_t ws_size,
                              hipStream_t stream) {
  (void)in_sizes; (void)n_in; (void)out_size; (void)ws_size;
  const float* x     = (const float*)d_in[0];
  const float* llam  = (const float*)d_in[1];
  const float* B_w   = (const float*)d_in[2];
  const float* C_w   = (const float*)d_in[3];
  const float* Dskip = (const float*)d_in[4];
  const float* ssm_w = (const float*)d_in[5];
  const float* ffn_w = (const float*)d_in[6];
  const float* w1    = (const float*)d_in[7];
  const float* w2    = (const float*)d_in[8];
  const float* w3    = (const float*)d_in[9];
  float* out = (float*)d_out;                 // also serves as x2 residual buffer

  char* ws = (char*)d_ws;
  size_t off = 0;
  auto alloc = [&](size_t bytes) -> char* {
    char* p = ws + off;
    off += (bytes + 255) & ~(size_t)255;
    return p;
  };
  u16*   u_bf  = (u16*)  alloc((size_t)M_ROWS * D_MODEL * 2);   // u, later z
  float* Bu    = (float*)alloc((size_t)M_ROWS * N_STATE * 4);
  u16*   hs_bf = (u16*)  alloc((size_t)M_ROWS * N_STATE * 2);
  float* csum  = (float*)alloc((size_t)B_SZ * 64 * N_STATE * 4);
  float* carry = (float*)alloc((size_t)B_SZ * 64 * N_STATE * 4);
  u16*   g_bf  = (u16*)  alloc((size_t)M_ROWS * D_FF_P * 2);
  u16*   Bw_bf = (u16*)  alloc((size_t)N_STATE * D_MODEL * 2);
  u16*   Cw_bf = (u16*)  alloc((size_t)D_MODEL * N_STATE * 2);
  u16*   w1_bf = (u16*)  alloc((size_t)D_FF_P * D_MODEL * 2);
  u16*   w3_bf = (u16*)  alloc((size_t)D_FF_P * D_MODEL * 2);
  u16*   w2_bf = (u16*)  alloc((size_t)D_MODEL * D_FF_P * 2);

  auto cvt = [&](const float* s, u16* d, int R, int K, int Kp, int Rp) {
    int total = Rp * Kp;
    k_cvt_bf16<<<(total + 255) / 256, 256, 0, stream>>>(s, d, R, K, Kp, total);
  };
  // per-launch weight conversion to bf16, with D_FF->2752 zero pad
  cvt(B_w, Bw_bf, N_STATE, D_MODEL, D_MODEL, N_STATE);
  cvt(C_w, Cw_bf, D_MODEL, N_STATE, N_STATE, D_MODEL);
  cvt(w1,  w1_bf, D_FF,    D_MODEL, D_MODEL, D_FF_P);
  cvt(w3,  w3_bf, D_FF,    D_MODEL, D_MODEL, D_FF_P);
  cvt(w2,  w2_bf, D_MODEL, D_FF,    D_FF_P,  D_MODEL);

  // 1) u = rmsnorm(x) -> bf16
  k_rmsnorm_bf16<<<M_ROWS, 256, 0, stream>>>(x, ssm_w, u_bf);

  // 2) Bu = u @ B_w^T   [16384 x 256], K=1024
  k_gemm_bf16<0, D_MODEL, D_MODEL, N_STATE>
      <<<dim3(N_STATE / 64, M_ROWS / 128), 256, 0, stream>>>(
      u_bf, Bw_bf, nullptr, Bu, nullptr, D_MODEL, nullptr, nullptr, nullptr);

  // 3) chunked scan -> hs (bf16)
  k_scan_chunk<<<(B_SZ * 64 * N_STATE) / 256, 256, 0, stream>>>(Bu, llam, csum);
  k_scan_carry<<<(B_SZ * N_STATE) / 256, 256, 0, stream>>>(llam, csum, carry);
  k_scan_out<<<(B_SZ * 64 * N_STATE) / 256, 256, 0, stream>>>(Bu, llam, carry, hs_bf);

  // 4) out(x2) = x + hs @ C_w^T + D_skip*u   [16384 x 1024], K=256
  k_gemm_bf16<1, N_STATE, N_STATE, D_MODEL>
      <<<dim3(D_MODEL / 64, M_ROWS / 128), 256, 0, stream>>>(
      hs_bf, Cw_bf, nullptr, out, nullptr, N_STATE, x, u_bf, Dskip);

  // 5) z = rmsnorm(x2) -> bf16 (reuse u buffer)
  k_rmsnorm_bf16<<<M_ROWS, 256, 0, stream>>>(out, ffn_w, u_bf);

  // 6) g = bf16(silu(z@w1^T) * (z@w3^T))   [16384 x 2752], K=1024, fused dual GEMM
  k_gemm_bf16<3, D_MODEL, D_MODEL, D_FF_P>
      <<<dim3(D_FF_P / 64, M_ROWS / 128), 256, 0, stream>>>(
      u_bf, w1_bf, w3_bf, nullptr, g_bf, D_MODEL, nullptr, nullptr, nullptr);

  // 7) out = x2 + g @ w2^T   [16384 x 1024], K=2752
  k_gemm_bf16<2, D_FF_P, D_FF_P, D_MODEL>
      <<<dim3(D_MODEL / 64, M_ROWS / 128), 256, 0, stream>>>(
      g_bf, w2_bf, nullptr, out, nullptr, D_FF_P, out, nullptr, nullptr);
}
